// DiT_11536282157395
// MI455X (gfx1250) — matrix operand order repaired and checked by reading, not compiled, not run on hardware
//
#include <hip/hip_runtime.h>
#include <cstdint>

#define DEV __device__ __forceinline__

constexpr int S = 1024, D = 1024, H = 16, DH = 64, L = 4, DFF = 4096;
constexpr int TPF = 8;

typedef __attribute__((ext_vector_type(16))) __bf16 bf16x16;
typedef __attribute__((ext_vector_type(8)))  __bf16 bf16x8;
typedef __attribute__((ext_vector_type(8)))  float  f32x8;

union Frag {
  bf16x16 v;
  bf16x8  h[2];
};

DEV unsigned short f2bf(float f) {
  uint32_t u = __float_as_uint(f);
  uint32_t r = (u + 0x7FFFu + ((u >> 16) & 1u)) >> 16;
  return (unsigned short)r;
}

DEV f32x8 wmma_bf16(Frag a, Frag b, f32x8 c) {
  return __builtin_amdgcn_wmma_f32_16x16x32_bf16(false, a.v, false, b.v,
                                                 (short)0, c, false, false);
}

// ---- CDNA5 async global->LDS copy (ASYNCcnt-tracked, CDNA5 ISA) ------
DEV void async_ld_b128(const void* gptr, void* lds_generic) {
  uint32_t lds_off = (uint32_t)(uintptr_t)lds_generic;  // flat-LDS: addr[31:0]
  asm volatile("global_load_async_to_lds_b128 %0, %1, off"
               :: "v"(lds_off), "v"((uint64_t)(uintptr_t)gptr)
               : "memory");
}
DEV void wait_async0() {
  asm volatile("s_wait_asynccnt 0x0" ::: "memory");
}

// ---- CDNA5 Tensor Data Mover: 2D tile DMA global->LDS (CDNA5 ISA/8) --
#if defined(__HIP_DEVICE_COMPILE__) && __has_builtin(__builtin_amdgcn_tensor_load_to_lds)
#define HAVE_TDM 1
#else
#define HAVE_TDM 0
#endif

#if HAVE_TDM
typedef __attribute__((ext_vector_type(4))) unsigned int u32x4;
typedef __attribute__((ext_vector_type(8))) int i32x8;
typedef __attribute__((ext_vector_type(4))) int i32x4;

// Load a [rows x 32] bf16 tile (row stride = strideK elements) packed into LDS.
// D# built per ISA: group0 = {count/type/lds/global addr}, group1 = dims/strides.
DEV void tdm_load_2d(uint32_t lds_off, const void* gptr, uint32_t strideK, uint32_t rows) {
  uint64_t ga = (uint64_t)(uintptr_t)gptr;
  u32x4 g0;
  g0[0] = 1u;                                            // count=1 (valid, user mode)
  g0[1] = lds_off;                                       // lds_addr (bytes)
  g0[2] = (uint32_t)ga;                                  // global_addr[31:0]
  g0[3] = ((uint32_t)(ga >> 32) & 0x1FFFFFFu) | (2u << 30);  // addr[56:32] | type=2
  i32x8 g1;
  g1[0] = (int)(1u << 16);        // workgroup_mask=0, data_size=1 (2 bytes)
  g1[1] = (int)(32u << 16);       // tensor_dim0[15:0]=32 at bits[63:48]
  g1[2] = (int)(rows << 16);      // tensor_dim0 hi=0 | tensor_dim1[15:0]=rows
  g1[3] = (int)(32u << 16);       // tensor_dim1 hi=0 | tile_dim0=32
  g1[4] = (int)rows;              // tile_dim1=rows | tile_dim2=0
  g1[5] = (int)strideK;           // tensor_dim0_stride[31:0]
  g1[6] = 0;                      // stride0 hi | tensor_dim1_stride lo
  g1[7] = 0;
  i32x4 z4 = (i32x4)0;
#if __clang_major__ >= 23
  i32x8 z8 = (i32x8)0;
  __builtin_amdgcn_tensor_load_to_lds(g0, g1, z4, z4, z8, 0);
#else
  __builtin_amdgcn_tensor_load_to_lds(g0, g1, z4, z4, 0);
#endif
}
#endif

// ---------------- modulation GEMVs: cond @ {ada1,gate1,ada2,gate2} ----------
__global__ void k_modulate(const float* __restrict__ cond,
                           const float* __restrict__ ada1_w, const float* __restrict__ ada1_b,
                           const float* __restrict__ gate1_w, const float* __restrict__ gate1_b,
                           const float* __restrict__ ada2_w, const float* __restrict__ ada2_b,
                           const float* __restrict__ gate2_w, const float* __restrict__ gate2_b,
                           float* __restrict__ mod) {
  int n = blockIdx.x * blockDim.x + threadIdx.x;   // 0..6143
  const float* w; const float* b; int col, N;
  if (n < 2048)      { w = ada1_w;  b = ada1_b;  col = n;        N = 2048; }
  else if (n < 3072) { w = gate1_w; b = gate1_b; col = n - 2048; N = 1024; }
  else if (n < 5120) { w = ada2_w;  b = ada2_b;  col = n - 3072; N = 2048; }
  else               { w = gate2_w; b = gate2_b; col = n - 5120; N = 1024; }
  float acc = 0.f;
  #pragma unroll 4
  for (int k = 0; k < D; ++k) acc += cond[k] * w[(size_t)k * N + col];
  mod[n] = acc + b[col];
}

// ---------------- fp32 [K][N] -> bf16 [N][K] transpose/convert --------------
__global__ void k_convert_w(const float* __restrict__ W,
                            unsigned short* __restrict__ Wt, int K, int N) {
  __shared__ float tile[32][33];
  int tx = threadIdx.x, ty = threadIdx.y;           // 32 x 8
  int n0 = blockIdx.x * 32, k0 = blockIdx.y * 32;
  #pragma unroll
  for (int j = 0; j < 4; ++j) {
    int k = k0 + ty + j * 8;
    tile[ty + j * 8][tx] = W[(size_t)k * N + n0 + tx];
  }
  __syncthreads();
  #pragma unroll
  for (int j = 0; j < 4; ++j) {
    int n = n0 + ty + j * 8;
    Wt[(size_t)n * K + k0 + tx] = f2bf(tile[tx][ty + j * 8]);
  }
}

// ---------------- layernorm + adaLN modulate -> bf16 ------------------------
__global__ void k_ln_mod(const float* __restrict__ x, const float* __restrict__ shift,
                         const float* __restrict__ scale, unsigned short* __restrict__ hb) {
  __shared__ float red[256];
  int row = blockIdx.x, tid = threadIdx.x;
  float v[4];
  float s = 0.f;
  #pragma unroll
  for (int t = 0; t < 4; ++t) { v[t] = x[(size_t)row * D + tid + t * 256]; s += v[t]; }
  red[tid] = s; __syncthreads();
  for (int o = 128; o > 0; o >>= 1) { if (tid < o) red[tid] += red[tid + o]; __syncthreads(); }
  float mean = red[0] * (1.f / D); __syncthreads();
  float s2 = 0.f;
  #pragma unroll
  for (int t = 0; t < 4; ++t) { float d = v[t] - mean; s2 += d * d; }
  red[tid] = s2; __syncthreads();
  for (int o = 128; o > 0; o >>= 1) { if (tid < o) red[tid] += red[tid + o]; __syncthreads(); }
  float rinv = rsqrtf(red[0] * (1.f / D) + 1e-5f);
  #pragma unroll
  for (int t = 0; t < 4; ++t) {
    int c = tid + t * 256;
    float y = (v[t] - mean) * rinv * (1.f + scale[c]) + shift[c];
    hb[(size_t)row * D + c] = f2bf(y);
  }
}

// ---------------- WMMA GEMM: C = A[M][K] @ Bt[N][K]^T + bias ----------------
// Tiles staged by the Tensor Data Mover (one wave issues D# pair per K-step),
// falling back to per-lane async global->LDS copies. Double-buffered.
// MODE 0: store f32    MODE 1: x += (C)*gate    MODE 2: gelu(C) -> bf16
template <int MODE>
__global__ __launch_bounds__(256) void k_gemm(
    const unsigned short* __restrict__ A, const unsigned short* __restrict__ Bt,
    const float* __restrict__ bias, int K, int N,
    float* __restrict__ outF, unsigned short* __restrict__ outB,
    const float* __restrict__ gate, float* __restrict__ xres) {
  __shared__ alignas(32) unsigned short As[2][64][32];
  __shared__ alignas(32) unsigned short Bs[2][128][32];
  int tid = threadIdx.x;
  int m0 = blockIdx.y * 64, n0 = blockIdx.x * 128;
  int lane = tid & 31, wid = tid >> 5;
  int wm = wid >> 2, wn = wid & 3;
  int lh = lane >> 4, ln = lane & 15;

#if HAVE_TDM
  auto stage = [&](int buf, int kb) {
    if (wid == 0) {   // TDM issues per-wave; EXEC ignored
      tdm_load_2d((uint32_t)(uintptr_t)&As[buf][0][0],
                  A + (size_t)m0 * K + kb, (uint32_t)K, 64u);
      tdm_load_2d((uint32_t)(uintptr_t)&Bs[buf][0][0],
                  Bt + (size_t)n0 * K + kb, (uint32_t)K, 128u);
    }
  };
  auto wait_stage = [&]() { __builtin_amdgcn_s_wait_tensorcnt(0); };
#else
  int ar = tid >> 2, ac = (tid & 3) * 8;   // staging coords (row, half-offset)
  const unsigned short* Ab  = A  + (size_t)(m0 + ar) * K + ac;
  const unsigned short* Bb0 = Bt + (size_t)(n0 + ar) * K + ac;
  const unsigned short* Bb1 = Bt + (size_t)(n0 + ar + 64) * K + ac;
  auto stage = [&](int buf, int kb) {
    async_ld_b128(Ab + kb,  &As[buf][ar][ac]);
    async_ld_b128(Bb0 + kb, &Bs[buf][ar][ac]);
    async_ld_b128(Bb1 + kb, &Bs[buf][ar + 64][ac]);
  };
  auto wait_stage = [&]() { wait_async0(); };
#endif

  stage(0, 0);
  wait_stage();
  __syncthreads();

  f32x8 acc[2][2] = {};
  int KT = K / 32;
  for (int kt = 0; kt < KT; ++kt) {
    int buf = kt & 1;
    if (kt + 1 < KT) stage((kt + 1) & 1, (kt + 1) * 32);   // DMA into other buffer

    Frag fa[2], fb[2];
    #pragma unroll
    for (int tm = 0; tm < 2; ++tm) {
      const unsigned short* p = &As[buf][wm * 32 + tm * 16 + ln][lh * 8];
      fa[tm].h[0] = *(const bf16x8*)p;
      fa[tm].h[1] = *(const bf16x8*)(p + 16);
    }
    #pragma unroll
    for (int tn = 0; tn < 2; ++tn) {
      const unsigned short* p = &Bs[buf][wn * 32 + tn * 16 + ln][lh * 8];
      fb[tn].h[0] = *(const bf16x8*)p;
      fb[tn].h[1] = *(const bf16x8*)(p + 16);
    }
    #pragma unroll
    for (int tm = 0; tm < 2; ++tm)
      #pragma unroll
      for (int tn = 0; tn < 2; ++tn)
        acc[tm][tn] = wmma_bf16(fa[tm], fb[tn], acc[tm][tn]);

    if (kt + 1 < KT) wait_stage();
    __syncthreads();
  }

  #pragma unroll
  for (int tm = 0; tm < 2; ++tm)
    #pragma unroll
    for (int tn = 0; tn < 2; ++tn)
      #pragma unroll
      for (int j = 0; j < 8; ++j) {
        int m = m0 + wm * 32 + tm * 16 + j + lh * 8;
        int n = n0 + wn * 32 + tn * 16 + ln;
        float v = acc[tm][tn][j] + bias[n];
        if (MODE == 0) {
          outF[(size_t)m * N + n] = v;
        } else if (MODE == 1) {
          xres[(size_t)m * D + n] += v * gate[n];
        } else {
          float g = 0.5f * v * (1.f + tanhf(0.7978845608f * (v + 0.044715f * v * v * v)));
          outB[(size_t)m * N + n] = f2bf(g);
        }
      }
}

// ---------------- per-(head,token) RMSNorm + RoPE, V transpose --------------
__global__ void k_qkv_post(const float* __restrict__ qkv, unsigned short* __restrict__ qh,
                           unsigned short* __restrict__ kh, unsigned short* __restrict__ vt) {
  int g = blockIdx.x * 8 + (threadIdx.x >> 5);
  int lane = threadIdx.x & 31;
  int s = g & (S - 1);
  int h = g >> 10;
  const float* base = qkv + (size_t)s * 3 * D + h * DH;
  float q0 = base[lane],         q1 = base[lane + 32];
  float k0 = base[D + lane],     k1 = base[D + lane + 32];
  float v0 = base[2 * D + lane], v1 = base[2 * D + lane + 32];
  float sq = q0 * q0 + q1 * q1, sk = k0 * k0 + k1 * k1;
  #pragma unroll
  for (int o = 16; o > 0; o >>= 1) { sq += __shfl_xor(sq, o, 32); sk += __shfl_xor(sk, o, 32); }
  float rq = rsqrtf(sq * (1.f / 64.f) + 1e-6f), rk = rsqrtf(sk * (1.f / 64.f) + 1e-6f);
  q0 *= rq; q1 *= rq; k0 *= rk; k1 *= rk;
  float inv = __powf(10000.f, -(float)lane / 32.f);
  float ang = (float)s * inv;
  float c = __cosf(ang), sn = __sinf(ang);
  float qa = q0 * c - q1 * sn, qb = q0 * sn + q1 * c;
  float ka = k0 * c - k1 * sn, kb = k0 * sn + k1 * c;
  size_t qo = ((size_t)h * S + s) * DH;
  qh[qo + lane] = f2bf(qa); qh[qo + 32 + lane] = f2bf(qb);
  kh[qo + lane] = f2bf(ka); kh[qo + 32 + lane] = f2bf(kb);
  vt[((size_t)h * DH + lane) * S + s]      = f2bf(v0);
  vt[((size_t)h * DH + lane + 32) * S + s] = f2bf(v1);
}

// ---------------- flash attention: wave per (head, 16-query block) ----------
__global__ __launch_bounds__(128) void k_attn(
    const unsigned short* __restrict__ qh, const unsigned short* __restrict__ kh,
    const unsigned short* __restrict__ vt, unsigned short* __restrict__ ao, int window) {
  __shared__ alignas(32) unsigned short Pl[4][16][32];
  int wid = threadIdx.x >> 5, lane = threadIdx.x & 31;
  int lh = lane >> 4, ln = lane & 15;
  int task = blockIdx.x * 4 + wid;
  int qb = task >> 4;
  int h = task & 15;

  Frag fq[2];
  {
    const unsigned short* qp = qh + ((size_t)h * S + qb * 16 + ln) * DH;
    #pragma unroll
    for (int c = 0; c < 2; ++c) {
      const unsigned short* p = qp + c * 32 + lh * 8;
      fq[c].h[0] = *(const bf16x8*)p;
      fq[c].h[1] = *(const bf16x8*)(p + 16);
    }
  }
  float mrow[8], lrow[8];
  f32x8 oacc[4] = {};
  #pragma unroll
  for (int j = 0; j < 8; ++j) { mrow[j] = -3.0e38f; lrow[j] = 0.f; }

  int kend = ((qb + 1) * 16 - 1) >> 5;     // inclusive, 32-key chunks (causal frame limit)
  int kstart = 0;
  if (window * TPF < S) {
    int minkey = qb * 16 - (window - 1) * TPF;
    kstart = minkey > 0 ? (minkey >> 5) : 0;
  }

  for (int kc = kstart; kc <= kend; ++kc) {
    int j0 = kc * 32;
    float sv[2][8];
    #pragma unroll
    for (int t = 0; t < 2; ++t) {
      Frag fk0, fk1;
      const unsigned short* kp = kh + ((size_t)h * S + j0 + t * 16 + ln) * DH + lh * 8;
      fk0.h[0] = *(const bf16x8*)kp;        fk0.h[1] = *(const bf16x8*)(kp + 16);
      fk1.h[0] = *(const bf16x8*)(kp + 32); fk1.h[1] = *(const bf16x8*)(kp + 48);
      f32x8 sc = {};
      sc = wmma_bf16(fq[0], fk0, sc);
      sc = wmma_bf16(fq[1], fk1, sc);
      int fk = (j0 + t * 16 + ln) >> 3;
      #pragma unroll
      for (int j = 0; j < 8; ++j) {
        int fqr = (qb * 16 + j + lh * 8) >> 3;
        bool okm = (fk <= fqr) && (fqr - fk < window);
        sv[t][j] = okm ? sc[j] * 0.125f : -1e30f;
      }
    }
    float sf[8];
    #pragma unroll
    for (int j = 0; j < 8; ++j) {
      float cm = fmaxf(sv[0][j], sv[1][j]);
      #pragma unroll
      for (int o = 8; o > 0; o >>= 1) cm = fmaxf(cm, __shfl_xor(cm, o, 32));
      float nm = fmaxf(mrow[j], cm);
      sf[j] = __expf(mrow[j] - nm);
      float p0 = sv[0][j] > -1e29f ? __expf(sv[0][j] - nm) : 0.f;
      float p1 = sv[1][j] > -1e29f ? __expf(sv[1][j] - nm) : 0.f;
      float rs = p0 + p1;
      #pragma unroll
      for (int o = 8; o > 0; o >>= 1) rs += __shfl_xor(rs, o, 32);
      lrow[j] = lrow[j] * sf[j] + rs;
      mrow[j] = nm;
      Pl[wid][j + lh * 8][ln]      = f2bf(p0);
      Pl[wid][j + lh * 8][16 + ln] = f2bf(p1);
    }
    #pragma unroll
    for (int f = 0; f < 4; ++f)
      #pragma unroll
      for (int j = 0; j < 8; ++j) oacc[f][j] *= sf[j];

    Frag fp;
    {
      const unsigned short* pp = &Pl[wid][ln][lh * 8];
      fp.h[0] = *(const bf16x8*)pp;
      fp.h[1] = *(const bf16x8*)(pp + 16);
    }
    #pragma unroll
    for (int f = 0; f < 4; ++f) {
      Frag fv;
      const unsigned short* vp = vt + ((size_t)h * DH + f * 16 + ln) * S + j0 + lh * 8;
      fv.h[0] = *(const bf16x8*)vp;
      fv.h[1] = *(const bf16x8*)(vp + 16);
      oacc[f] = wmma_bf16(fp, fv, oacc[f]);
    }
  }

  #pragma unroll
  for (int j = 0; j < 8; ++j) lrow[j] = 1.f / lrow[j];
  int mbase = qb * 16 + lh * 8;
  #pragma unroll
  for (int f = 0; f < 4; ++f) {
    int col = h * DH + f * 16 + ln;
    #pragma unroll
    for (int j = 0; j < 8; ++j)
      ao[(size_t)(mbase + j) * D + col] = f2bf(oacc[f][j] * lrow[j]);
  }
}

// ---------------------------------------------------------------------------
extern "C" void kernel_launch(void* const* d_in, const int* in_sizes, int n_in,
                              void* d_out, int out_size, void* d_ws, size_t ws_size,
                              hipStream_t stream) {
  const float* x_in    = (const float*)d_in[0];
  const float* cond    = (const float*)d_in[1];
  const float* qkv_w   = (const float*)d_in[2];
  const float* qkv_b   = (const float*)d_in[3];
  const float* out_w   = (const float*)d_in[4];
  const float* out_b   = (const float*)d_in[5];
  const float* mlp_w1  = (const float*)d_in[6];
  const float* mlp_b1  = (const float*)d_in[7];
  const float* mlp_w2  = (const float*)d_in[8];
  const float* mlp_b2  = (const float*)d_in[9];
  const float* ada1_w  = (const float*)d_in[10];
  const float* ada1_b  = (const float*)d_in[11];
  const float* gate1_w = (const float*)d_in[12];
  const float* gate1_b = (const float*)d_in[13];
  const float* ada2_w  = (const float*)d_in[14];
  const float* ada2_b  = (const float*)d_in[15];
  const float* gate2_w = (const float*)d_in[16];
  const float* gate2_b = (const float*)d_in[17];
  (void)in_sizes; (void)n_in; (void)out_size; (void)ws_size;

  float* x = (float*)d_out;
  hipMemcpyAsync(x, x_in, (size_t)S * D * sizeof(float), hipMemcpyDeviceToDevice, stream);

  char* w = (char*)d_ws;
  auto alloc = [&](size_t bytes) { char* p = w; w += (bytes + 255) & ~(size_t)255; return p; };
  unsigned short* hb     = (unsigned short*)alloc((size_t)S * D * 2);
  float*          qkv    = (float*)alloc((size_t)S * 3 * D * 4);
  unsigned short* qh     = (unsigned short*)alloc((size_t)S * D * 2);
  unsigned short* kh     = (unsigned short*)alloc((size_t)S * D * 2);
  unsigned short* vt     = (unsigned short*)alloc((size_t)S * D * 2);
  unsigned short* ao     = (unsigned short*)alloc((size_t)S * D * 2);
  unsigned short* h1     = (unsigned short*)alloc((size_t)S * DFF * 2);
  unsigned short* qkvw_t = (unsigned short*)alloc((size_t)3 * D * D * 2);
  unsigned short* outw_t = (unsigned short*)alloc((size_t)D * D * 2);
  unsigned short* w1t    = (unsigned short*)alloc((size_t)DFF * D * 2);
  unsigned short* w2t    = (unsigned short*)alloc((size_t)D * DFF * 2);
  float*          mod    = (float*)alloc(6144 * 4);

  for (int i = 0; i < L; ++i) {
    int window = (i % 4 == 0) ? 128 : 8;

    k_modulate<<<24, 256, 0, stream>>>(cond,
        ada1_w + (size_t)i * D * 2 * D, ada1_b + (size_t)i * 2 * D,
        gate1_w + (size_t)i * D * D,    gate1_b + (size_t)i * D,
        ada2_w + (size_t)i * D * 2 * D, ada2_b + (size_t)i * 2 * D,
        gate2_w + (size_t)i * D * D,    gate2_b + (size_t)i * D, mod);

    k_convert_w<<<dim3(3 * D / 32, D / 32), dim3(32, 8), 0, stream>>>(
        qkv_w + (size_t)i * D * 3 * D, qkvw_t, D, 3 * D);
    k_convert_w<<<dim3(D / 32, D / 32), dim3(32, 8), 0, stream>>>(
        out_w + (size_t)i * D * D, outw_t, D, D);
    k_convert_w<<<dim3(DFF / 32, D / 32), dim3(32, 8), 0, stream>>>(
        mlp_w1 + (size_t)i * D * DFF, w1t, D, DFF);
    k_convert_w<<<dim3(D / 32, DFF / 32), dim3(32, 8), 0, stream>>>(
        mlp_w2 + (size_t)i * DFF * D, w2t, DFF, D);

    k_ln_mod<<<S, 256, 0, stream>>>(x, mod + 0, mod + D, hb);
    k_gemm<0><<<dim3(3 * D / 128, S / 64), 256, 0, stream>>>(
        hb, qkvw_t, qkv_b + (size_t)i * 3 * D, D, 3 * D, qkv, nullptr, nullptr, nullptr);
    k_qkv_post<<<S * H / 8, 256, 0, stream>>>(qkv, qh, kh, vt);
    k_attn<<<(S / 16) * H / 4, 128, 0, stream>>>(qh, kh, vt, ao, window);
    k_gemm<1><<<dim3(D / 128, S / 64), 256, 0, stream>>>(
        ao, outw_t, out_b + (size_t)i * D, D, D, nullptr, nullptr, mod + 2048, x);

    k_ln_mod<<<S, 256, 0, stream>>>(x, mod + 3072, mod + 3072 + D, hb);
    k_gemm<2><<<dim3(DFF / 128, S / 64), 256, 0, stream>>>(
        hb, w1t, mlp_b1 + (size_t)i * DFF, D, DFF, nullptr, h1, nullptr, nullptr);
    k_gemm<1><<<dim3(D / 128, S / 64), 256, 0, stream>>>(
        h1, w2t, mlp_b2 + (size_t)i * D, DFF, D, nullptr, nullptr, mod + 5120, x);
  }
}
